// PatientGNN_65249143161469
// MI455X (gfx1250) — matrix operand order repaired and checked by reading, not compiled, not run on hardware
//
#include <hip/hip_runtime.h>
#include <hip/hip_bf16.h>
#include <stdint.h>

// ---------------------------------------------------------------------------
// CDNA5 (gfx1250) implementation of the PatientGNN reference.
//  - Dense GEMMs: templated, fully-unrolled v_wmma_f32_16x16x32_bf16
//    (f32 -> bf16 RNE inputs, f32 accumulate, float4 A-fragment loads).
//  - Graph propagation: L2-resident gathers + f32 global atomics
//    (feature working set < 192MB L2 on MI455X).
//  - GATv2 segment-max: order-preserving uint mapping + atomicMax.
//  - Reference's edge-MLP (_new_ew) is dead code -> skipped.
// ---------------------------------------------------------------------------

typedef __attribute__((ext_vector_type(16))) __bf16         v16bf;
typedef __attribute__((ext_vector_type(16))) unsigned short v16us;
typedef __attribute__((ext_vector_type(8)))  float          v8f;

__device__ __forceinline__ unsigned short f2bf(float f) {
    unsigned u = __float_as_uint(f);
    unsigned r = u + 0x7FFFu + ((u >> 16) & 1u);   // round-to-nearest-even
    return (unsigned short)(r >> 16);
}

__device__ __forceinline__ float wave_sum(float v) {
    #pragma unroll
    for (int off = 16; off > 0; off >>= 1) v += __shfl_xor(v, off, 32);
    return v;
}

// order-preserving float <-> uint mapping (for atomic segment-max)
__device__ __forceinline__ unsigned f2ord(float f) {
    unsigned u = __float_as_uint(f);
    return (u & 0x80000000u) ? ~u : (u | 0x80000000u);
}
__device__ __forceinline__ float ord2f(unsigned u) {
    unsigned v = (u & 0x80000000u) ? (u & 0x7FFFFFFFu) : ~u;
    return __uint_as_float(v);
}

// ---------------------------------------------------------------------------
// WMMA GEMM: C[M,NOUT] = A[M,K] @ W[K,NOUT] (+bias if BIAS) (+=C if ACC)
// One wave per 16x16 tile; K fully unrolled in steps of 32 (bf16 WMMA).
// ---------------------------------------------------------------------------
template <int K, int NOUT, bool ACC, bool BIAS>
__global__ void gemm_bf16_wmma_kernel(const float* __restrict__ A,
                                      const float* __restrict__ W,
                                      const float* __restrict__ bias,
                                      float* __restrict__ C, int M) {
    constexpr int tilesN = NOUT >> 4;
    int wid    = blockIdx.x * (blockDim.x >> 5) + (threadIdx.x >> 5);
    int tilesM = (M + 15) >> 4;
    if (wid >= tilesM * tilesN) return;
    int tm = wid / tilesN, tn = wid % tilesN;
    int lane = threadIdx.x & 31;
    int half = lane >> 4;      // 16-lane group
    int q    = lane & 15;      // row (A) / column (B,C) within tile
    int am = tm * 16 + q; if (am >= M) am = M - 1;   // clamp (masked on store)
    const float* arow = A + (long long)am * K;
    const int ncol = tn * 16 + q;

    v8f acc = {};
    #pragma unroll
    for (int k0 = 0; k0 < K; k0 += 32) {
        // ---- A fragment: 16x32 bf16 layout ----
        // lane's vector j in [0,8)  -> K = k0 + half*8 + j      (8 contiguous f32)
        // lane's vector j in [8,16) -> K = k0 + 16 + half*8 + j-8 (8 contiguous f32)
        float4 a0 = *(const float4*)(arow + k0 + half * 8);
        float4 a1 = *(const float4*)(arow + k0 + half * 8 + 4);
        float4 a2 = *(const float4*)(arow + k0 + 16 + half * 8);
        float4 a3 = *(const float4*)(arow + k0 + 16 + half * 8 + 4);
        v16us au;
        au[0]  = f2bf(a0.x); au[1]  = f2bf(a0.y); au[2]  = f2bf(a0.z); au[3]  = f2bf(a0.w);
        au[4]  = f2bf(a1.x); au[5]  = f2bf(a1.y); au[6]  = f2bf(a1.z); au[7]  = f2bf(a1.w);
        au[8]  = f2bf(a2.x); au[9]  = f2bf(a2.y); au[10] = f2bf(a2.z); au[11] = f2bf(a2.w);
        au[12] = f2bf(a3.x); au[13] = f2bf(a3.y); au[14] = f2bf(a3.z); au[15] = f2bf(a3.w);
        // ---- B fragment: same K order as A: j<8 -> K=k0+half*8+j, j>=8 -> K=k0+16+half*8+j-8 ----
        v16us bu;
        #pragma unroll
        for (int j = 0; j < 16; ++j)
            bu[j] = f2bf(W[(long long)(k0 + half * 8 + j + (j >= 8 ? 8 : 0)) * NOUT + ncol]);

        acc = __builtin_amdgcn_wmma_f32_16x16x32_bf16(
                  false, __builtin_bit_cast(v16bf, au),
                  false, __builtin_bit_cast(v16bf, bu),
                  (short)0, acc, false, false);
    }

    float bv = BIAS ? bias[ncol] : 0.0f;
    float* crow = C + (long long)(tm * 16 + half * 8) * NOUT + ncol;
    if (tm * 16 + 16 <= M) {           // full tile: unguarded epilogue
        #pragma unroll
        for (int r = 0; r < 8; ++r) {  // C: VGPR r -> M = half*8 + r
            float v = acc[r] + bv;
            if (ACC) v += crow[(long long)r * NOUT];
            crow[(long long)r * NOUT] = v;
        }
    } else {                           // tail tile
        #pragma unroll
        for (int r = 0; r < 8; ++r) {
            int m = tm * 16 + half * 8 + r;
            if (m < M) {
                float v = acc[r] + bv;
                if (ACC) v += crow[(long long)r * NOUT];
                crow[(long long)r * NOUT] = v;
            }
        }
    }
}

// ---------------------------------------------------------------------------
// Elementwise / graph kernels
// ---------------------------------------------------------------------------
__global__ void fill_kernel(float* p, float v, long long n) {
    long long t = (long long)blockIdx.x * blockDim.x + threadIdx.x;
    if (t < n) p[t] = v;
}

__global__ void deg_scatter_kernel(const int* __restrict__ dst,
                                   const float* __restrict__ ew,
                                   float* __restrict__ deg, int E) {
    int e = blockIdx.x * blockDim.x + threadIdx.x;
    if (e < E) atomicAdd(&deg[dst[e]], ew[e]);
}

__global__ void rsqrt_kernel(float* deg, int N) {
    int i = blockIdx.x * blockDim.x + threadIdx.x;
    if (i < N) { float d = deg[i]; deg[i] = (d > 0.0f) ? rsqrtf(d) : 0.0f; }
}

// gcn_out init with self-loop contribution: dinv[i]^2 * xw[i]
__global__ void gcn_self_kernel(const float* __restrict__ xw,
                                const float* __restrict__ dinv,
                                float* __restrict__ out, int N) {
    long long t = (long long)blockIdx.x * blockDim.x + threadIdx.x;
    if (t < (long long)N * 64) {
        int i = (int)(t >> 6);
        float di = dinv[i];
        out[t] = di * di * xw[t];
    }
}

__global__ void gcn_scatter_kernel(const int* __restrict__ src, const int* __restrict__ dst,
                                   const float* __restrict__ ew, const float* __restrict__ dinv,
                                   const float* __restrict__ xw, float* __restrict__ out, int E) {
    long long t = (long long)blockIdx.x * blockDim.x + threadIdx.x;
    if (t >= (long long)E * 64) return;
    int e = (int)(t >> 6), c = (int)(t & 63);
    int s = src[e], d = dst[e];
    float norm = dinv[s] * ew[e] * dinv[d];
    atomicAdd(&out[(long long)d * 64 + c], norm * xw[(long long)s * 64 + c]);
}

// out = relu(LayerNorm(in + pre_bias)) + residual      (one wave per node, HID=64)
__global__ void ln_relu_add_kernel(const float* __restrict__ in, const float* __restrict__ pre_bias,
                                   const float* __restrict__ g, const float* __restrict__ b,
                                   const float* __restrict__ residual, float* __restrict__ out, int N) {
    int wid = blockIdx.x * (blockDim.x >> 5) + (threadIdx.x >> 5);
    if (wid >= N) return;
    int lane = threadIdx.x & 31;
    long long base = (long long)wid * 64;
    float v0 = in[base + lane], v1 = in[base + lane + 32];
    if (pre_bias) { v0 += pre_bias[lane]; v1 += pre_bias[lane + 32]; }
    float mu = wave_sum(v0 + v1) * (1.0f / 64.0f);
    float d0 = v0 - mu, d1 = v1 - mu;
    float var = wave_sum(d0 * d0 + d1 * d1) * (1.0f / 64.0f);
    float rs = rsqrtf(var + 1e-5f);
    float y0 = fmaxf(d0 * rs * g[lane] + b[lane], 0.0f);
    float y1 = fmaxf(d1 * rs * g[lane + 32] + b[lane + 32], 0.0f);
    if (residual) { y0 += residual[base + lane]; y1 += residual[base + lane + 32]; }
    out[base + lane] = y0;
    out[base + lane + 32] = y1;
}

// GATv2 scores: one wave per (virtual) edge; 4 heads, C=64
__global__ void gat_score_kernel(const int* __restrict__ src, const int* __restrict__ dst,
                                 const float* __restrict__ xl, const float* __restrict__ xr,
                                 const float* __restrict__ att,
                                 float* __restrict__ score, unsigned* __restrict__ smax,
                                 int E, int N) {
    int wid = blockIdx.x * (blockDim.x >> 5) + (threadIdx.x >> 5);
    int EN = E + N;
    if (wid >= EN) return;
    int lane = threadIdx.x & 31;
    int s, d;
    if (wid < E) { s = src[wid]; d = dst[wid]; } else { s = d = wid - E; }
    const float* pl = xl + (long long)s * 256;
    const float* pr = xr + (long long)d * 256;
    float sc[4];
    #pragma unroll
    for (int h = 0; h < 4; ++h) {
        float p = 0.0f;
        #pragma unroll
        for (int cc = 0; cc < 2; ++cc) {
            int c = lane + cc * 32;
            float v = pl[h * 64 + c] + pr[h * 64 + c];
            v = (v > 0.0f) ? v : 0.2f * v;          // leaky_relu 0.2
            p += v * att[h * 64 + c];
        }
        sc[h] = wave_sum(p);
    }
    if (lane == 0) {
        long long so = (long long)wid * 4;
        #pragma unroll
        for (int h = 0; h < 4; ++h) {
            score[so + h] = sc[h];
            atomicMax(&smax[(long long)d * 4 + h], f2ord(sc[h]));
        }
    }
}

// ex = exp(score - smax[d]); score <- ex; den[d] += ex
__global__ void gat_exp_kernel(const int* __restrict__ dst,
                               const unsigned* __restrict__ smax,
                               float* __restrict__ score, float* __restrict__ den,
                               int E, int N) {
    long long t = (long long)blockIdx.x * blockDim.x + threadIdx.x;
    long long EN = (long long)E + N;
    if (t >= EN * 4) return;
    int e = (int)(t >> 2), h = (int)(t & 3);
    int d = (e < E) ? dst[e] : (e - E);
    float m = ord2f(smax[(long long)d * 4 + h]);
    float ex = __expf(score[t] - m);
    score[t] = ex;
    atomicAdd(&den[(long long)d * 4 + h], ex);
}

// acc[d, h, c] += (ex/den[d,h]) * xl[s, h, c]
__global__ void gat_accum_kernel(const int* __restrict__ src, const int* __restrict__ dst,
                                 const float* __restrict__ xl, const float* __restrict__ score,
                                 const float* __restrict__ den, float* __restrict__ acc,
                                 int E, int N) {
    long long t = (long long)blockIdx.x * blockDim.x + threadIdx.x;
    long long EN = (long long)E + N;
    if (t >= EN * 256) return;
    int e = (int)(t >> 8), idx = (int)(t & 255), h = idx >> 6;
    int s, d;
    if (e < E) { s = src[e]; d = dst[e]; } else { s = d = e - E; }
    float alpha = score[(long long)e * 4 + h] / den[(long long)d * 4 + h];
    atomicAdd(&acc[(long long)d * 256 + idx], alpha * xl[(long long)s * 256 + idx]);
}

// mean over heads
__global__ void gat_mean_kernel(const float* __restrict__ acc, float* __restrict__ out, int N) {
    long long t = (long long)blockIdx.x * blockDim.x + threadIdx.x;
    if (t >= (long long)N * 64) return;
    int i = (int)(t >> 6), c = (int)(t & 63);
    long long b = (long long)i * 256;
    out[t] = 0.25f * (acc[b + c] + acc[b + 64 + c] + acc[b + 128 + c] + acc[b + 192 + c]);
}

__global__ void sage_scatter_kernel(const int* __restrict__ src, const int* __restrict__ dst,
                                    const float* __restrict__ h2, float* __restrict__ nsum,
                                    float* __restrict__ cnt, int E) {
    long long t = (long long)blockIdx.x * blockDim.x + threadIdx.x;
    if (t >= (long long)E * 64) return;
    int e = (int)(t >> 6), c = (int)(t & 63);
    int s = src[e], d = dst[e];
    atomicAdd(&nsum[(long long)d * 64 + c], h2[(long long)s * 64 + c]);
    if (c == 0) atomicAdd(&cnt[d], 1.0f);
}

__global__ void sage_mean_kernel(float* __restrict__ nsum, const float* __restrict__ cnt, int N) {
    long long t = (long long)blockIdx.x * blockDim.x + threadIdx.x;
    if (t >= (long long)N * 64) return;
    int i = (int)(t >> 6);
    nsum[t] /= fmaxf(cnt[i], 1.0f);
}

// JK-cat + masked readout + 3 heads; one wave per node; output order: mort[N], hours[N], disc[N,4]
__global__ void heads_kernel(const float* __restrict__ h_init, const float* __restrict__ h2,
                             const float* __restrict__ h3, const unsigned char* __restrict__ mask,
                             const float* __restrict__ mW, const float* __restrict__ mB,
                             const float* __restrict__ hW, const float* __restrict__ hB,
                             const float* __restrict__ dW, const float* __restrict__ dB,
                             float* __restrict__ out, int N) {
    int wid = blockIdx.x * (blockDim.x >> 5) + (threadIdx.x >> 5);
    if (wid >= N) return;
    int lane = threadIdx.x & 31;
    float msk = mask[wid] ? 1.0f : 0.0f;
    float pm = 0, ph = 0, pd0 = 0, pd1 = 0, pd2 = 0, pd3 = 0;
    long long base = (long long)wid * 64;
    #pragma unroll
    for (int jj = 0; jj < 6; ++jj) {
        int j = lane + jj * 32;                         // 0..191
        float v;
        if (j < 64)        v = h_init[base + j];
        else if (j < 128)  v = h2[base + j - 64];
        else               v = h3[base + j - 128];
        v *= msk;
        pm  += v * mW[j];
        ph  += v * hW[j];
        pd0 += v * dW[j * 4 + 0];
        pd1 += v * dW[j * 4 + 1];
        pd2 += v * dW[j * 4 + 2];
        pd3 += v * dW[j * 4 + 3];
    }
    pm = wave_sum(pm); ph = wave_sum(ph);
    pd0 = wave_sum(pd0); pd1 = wave_sum(pd1); pd2 = wave_sum(pd2); pd3 = wave_sum(pd3);
    if (lane == 0) {
        out[wid]                      = pm + mB[0];
        out[(long long)N + wid]       = ph + hB[0];
        long long db = 2LL * N + (long long)wid * 4;
        out[db + 0] = pd0 + dB[0];
        out[db + 1] = pd1 + dB[1];
        out[db + 2] = pd2 + dB[2];
        out[db + 3] = pd3 + dB[3];
    }
}

// ---------------------------------------------------------------------------
extern "C" void kernel_launch(void* const* d_in, const int* in_sizes, int n_in,
                              void* d_out, int out_size, void* d_ws, size_t ws_size,
                              hipStream_t stream) {
    const float* x       = (const float*)d_in[0];
    const int*   ei      = (const int*)d_in[1];
    const float* ew      = (const float*)d_in[2];
    const unsigned char* mask = (const unsigned char*)d_in[3];
    const float* gcn_W   = (const float*)d_in[4];
    const float* gcn_b   = (const float*)d_in[5];
    const float* proj_W  = (const float*)d_in[6];
    const float* proj_b  = (const float*)d_in[7];
    const float* ln1_g   = (const float*)d_in[8];
    const float* ln1_b   = (const float*)d_in[9];
    const float* gat_Wl  = (const float*)d_in[10];
    const float* gat_Wr  = (const float*)d_in[11];
    const float* gat_att = (const float*)d_in[12];
    const float* gat_b   = (const float*)d_in[13];
    const float* ln2_g   = (const float*)d_in[14];
    const float* ln2_b   = (const float*)d_in[15];
    const float* sage_Wl = (const float*)d_in[16];
    const float* sage_bl = (const float*)d_in[17];
    const float* sage_Wr = (const float*)d_in[18];
    const float* ln3_g   = (const float*)d_in[19];
    const float* ln3_b   = (const float*)d_in[20];
    // d_in[21..24] = edge MLP params: dead code in the reference, skipped.
    const float* mort_W  = (const float*)d_in[25];
    const float* mort_b  = (const float*)d_in[26];
    const float* hours_W = (const float*)d_in[27];
    const float* hours_b = (const float*)d_in[28];
    const float* disc_W  = (const float*)d_in[29];
    const float* disc_b  = (const float*)d_in[30];

    const int N = in_sizes[0] / 128;
    const int E = in_sizes[1] / 2;
    const int* src = ei;
    const int* dst = ei + E;
    float* out = (float*)d_out;

    // ---- workspace carve ----
    size_t off = 0;
    auto carve = [&](size_t bytes) -> void* {
        void* p = (char*)d_ws + off;
        off += (bytes + 255) & ~(size_t)255;
        return p;
    };
    float*    h_init = (float*)carve((size_t)N * 64 * 4);
    float*    h2     = (float*)carve((size_t)N * 64 * 4);
    float*    h3     = (float*)carve((size_t)N * 64 * 4);
    float*    bufA   = (float*)carve((size_t)N * 64 * 4);   // xw / nsum / nmean
    float*    bufB   = (float*)carve((size_t)N * 64 * 4);   // proj / sage_out
    float*    bufC   = (float*)carve((size_t)N * 64 * 4);   // gcn_out / gat_mean
    float*    xl     = (float*)carve((size_t)N * 256 * 4);
    float*    xr     = (float*)carve((size_t)N * 256 * 4);  // reused as GAT accumulator
    float*    score  = (float*)carve(((size_t)E + N) * 4 * 4);
    unsigned* smax   = (unsigned*)carve((size_t)N * 4 * 4);
    float*    den    = (float*)carve((size_t)N * 4 * 4);
    float*    deg    = (float*)carve((size_t)N * 4);        // becomes dinv in place
    float*    cnt    = (float*)carve((size_t)N * 4);

    const dim3 blk(256);
    auto grid1 = [](long long n) { return dim3((unsigned)((n + 255) / 256)); };
    auto gridW = [](long long waves) { return dim3((unsigned)((waves + 7) / 8)); };  // 8 waves/block

    const long long tilesM = (N + 15) / 16;

    // ---- GCN ----
    fill_kernel<<<grid1(N), blk, 0, stream>>>(deg, 1.0f, N);         // self-loop weight 1
    deg_scatter_kernel<<<grid1(E), blk, 0, stream>>>(dst, ew, deg, E);
    rsqrt_kernel<<<grid1(N), blk, 0, stream>>>(deg, N);              // deg -> dinv
    gemm_bf16_wmma_kernel<128, 64, false, false>
        <<<gridW(tilesM * 4), blk, 0, stream>>>(x, gcn_W, nullptr, bufA, N);
    gemm_bf16_wmma_kernel<128, 64, false, true>
        <<<gridW(tilesM * 4), blk, 0, stream>>>(x, proj_W, proj_b, bufB, N);
    gcn_self_kernel<<<grid1((long long)N * 64), blk, 0, stream>>>(bufA, deg, bufC, N);
    gcn_scatter_kernel<<<grid1((long long)E * 64), blk, 0, stream>>>(src, dst, ew, deg, bufA, bufC, E);
    ln_relu_add_kernel<<<gridW(N), blk, 0, stream>>>(bufC, gcn_b, ln1_g, ln1_b, bufB, h_init, N);

    // ---- GATv2 ----
    gemm_bf16_wmma_kernel<64, 256, false, false>
        <<<gridW(tilesM * 16), blk, 0, stream>>>(h_init, gat_Wl, nullptr, xl, N);
    gemm_bf16_wmma_kernel<64, 256, false, false>
        <<<gridW(tilesM * 16), blk, 0, stream>>>(h_init, gat_Wr, nullptr, xr, N);
    hipMemsetAsync(smax, 0, (size_t)N * 4 * 4, stream);   // ordered-uint 0 == -inf
    hipMemsetAsync(den, 0, (size_t)N * 4 * 4, stream);
    gat_score_kernel<<<gridW((long long)E + N), blk, 0, stream>>>(src, dst, xl, xr, gat_att, score, smax, E, N);
    hipMemsetAsync(xr, 0, (size_t)N * 256 * 4, stream);   // xr dead -> reuse as accumulator
    gat_exp_kernel<<<grid1(((long long)E + N) * 4), blk, 0, stream>>>(dst, smax, score, den, E, N);
    gat_accum_kernel<<<grid1(((long long)E + N) * 256), blk, 0, stream>>>(src, dst, xl, score, den, xr, E, N);
    gat_mean_kernel<<<grid1((long long)N * 64), blk, 0, stream>>>(xr, bufC, N);
    ln_relu_add_kernel<<<gridW(N), blk, 0, stream>>>(bufC, gat_b, ln2_g, ln2_b, h_init, h2, N);

    // ---- SAGE ----
    hipMemsetAsync(bufA, 0, (size_t)N * 64 * 4, stream);
    hipMemsetAsync(cnt, 0, (size_t)N * 4, stream);
    sage_scatter_kernel<<<grid1((long long)E * 64), blk, 0, stream>>>(src, dst, h2, bufA, cnt, E);
    sage_mean_kernel<<<grid1((long long)N * 64), blk, 0, stream>>>(bufA, cnt, N);
    gemm_bf16_wmma_kernel<64, 64, false, true>
        <<<gridW(tilesM * 4), blk, 0, stream>>>(bufA, sage_Wl, sage_bl, bufB, N);
    gemm_bf16_wmma_kernel<64, 64, true, false>
        <<<gridW(tilesM * 4), blk, 0, stream>>>(h2, sage_Wr, nullptr, bufB, N);
    ln_relu_add_kernel<<<gridW(N), blk, 0, stream>>>(bufB, nullptr, ln3_g, ln3_b, h2, h3, N);

    // ---- heads ----
    heads_kernel<<<gridW(N), blk, 0, stream>>>(h_init, h2, h3, mask,
                                               mort_W, mort_b, hours_W, hours_b,
                                               disc_W, disc_b, out, N);
    (void)n_in; (void)out_size; (void)ws_size;
}
